// Attention_62139586838947
// MI455X (gfx1250) — compile-verified, matrix operand order checked by reading, not run on hardware
//
#include <hip/hip_runtime.h>
#include <hip/hip_bf16.h>

typedef __attribute__((ext_vector_type(16))) __bf16 v16bf;
typedef __attribute__((ext_vector_type(8)))  __bf16 v8bf;
typedef __attribute__((ext_vector_type(8)))  float  v8f;
typedef __attribute__((ext_vector_type(4)))  unsigned int u32x4;
typedef __attribute__((ext_vector_type(8)))  unsigned int u32x8;

#define BATCH 8
#define SEQ   1024
#define DIM   1024
#define HEADS 16
#define HDIM  64
#define BN    (BATCH * SEQ)   // 8192 rows

union FragU { v16bf v; v8bf h[2]; };

// ---------------------------------------------------------------------------
// Fragment load from row-major GLOBAL memory (ld in elements).
// Lane l -> row (l&15); elems 0..7 at cols col0+g*8, elems 8..15 at col0+16+g*8.
// ---------------------------------------------------------------------------
__device__ inline v16bf load_frag_g(const __bf16* __restrict__ p, int row0, int col0, int ld) {
    int lane = threadIdx.x & 31;
    int g = lane >> 4;
    const __bf16* base = p + (size_t)(row0 + (lane & 15)) * ld + col0 + g * 8;
    FragU u;
    u.h[0] = *(const v8bf*)(base);
    u.h[1] = *(const v8bf*)(base + 16);
    return u.v;
}

// ---------------------------------------------------------------------------
// Fragment load from an LDS tile written by TDM with padding: 16B pad inserted
// after every 64B stored. Logical row length = rowBytes; physical pitch =
// rowBytes + (rowBytes/64)*16. Each 16B run starts 16B-aligned and never
// crosses a 64B boundary, so runs stay contiguous.
// ---------------------------------------------------------------------------
__device__ inline v16bf load_frag_lds(const __bf16* p, int row0, int col0, int rowBytes) {
    int lane = threadIdx.x & 31;
    int g = lane >> 4;
    int pitch = rowBytes + ((rowBytes >> 6) << 4);
    int L0 = (col0 + g * 8) * 2;       // logical byte offset of elems 0..7
    int L1 = L0 + 32;                  // logical byte offset of elems 8..15
    const char* rowp = (const char*)p + (row0 + (lane & 15)) * pitch;
    FragU u;
    u.h[0] = *(const v8bf*)(rowp + L0 + ((L0 >> 6) << 4));
    u.h[1] = *(const v8bf*)(rowp + L1 + ((L1 >> 6) << 4));
    return u.v;
}

__device__ inline v8f wmma_bf16(v16bf a, v16bf b, v8f c) {
    return __builtin_amdgcn_wmma_f32_16x16x32_bf16(false, a, false, b, (short)0, c, false, false);
}

// Pin scheduling: `nds` DS-reads first, then `nmma` WMMAs (per region).
__device__ inline void sched_ds_then_wmma(void) {
    __builtin_amdgcn_sched_group_barrier(0x100, 10, 0);  // DS read x10
    __builtin_amdgcn_sched_group_barrier(0x008, 4, 0);   // MFMA/WMMA x4
}

// ---------------------------------------------------------------------------
// Tensor Data Mover: async 2D bf16 tile load Global -> LDS.
// D# group0 (4 SGPRs) + group1 (8 SGPRs) per CDNA5 ISA 8.3/8.4.
// data_size=2B; pad_enable: 16B pad (pad_amount code 3 = 4 DWORDs) every 64B
// (pad_interval code 3 = 16 DWORDs). Tracked by TENSORcnt.
// ---------------------------------------------------------------------------
__device__ inline void tdm_load_2d(unsigned lds_addr, const __bf16* gptr,
                                   unsigned tensor_d0, unsigned tensor_d1,
                                   unsigned tile_d0, unsigned tile_d1,
                                   unsigned stride0) {
    unsigned long long ga = (unsigned long long)(size_t)gptr;
    union { u32x4 v; unsigned u[4]; } g0;
    union { u32x8 v; unsigned u[8]; } g1;
    g0.u[0] = 1u;                                              // count=1 (valid D#)
    g0.u[1] = lds_addr;                                        // LDS byte address
    g0.u[2] = (unsigned)ga;                                    // global addr [31:0]
    g0.u[3] = (unsigned)((ga >> 32) & 0x01FFFFFFu) | (2u << 30); // addr[56:32] | type=2
    g1.u[0] = (1u << 16) | (1u << 20) | (3u << 22) | (3u << 25); // data_size=2B, pad 16B/64B
    g1.u[1] = (tensor_d0 & 0xFFFFu) << 16;                     // tensor_dim0[15:0]
    g1.u[2] = (tensor_d0 >> 16) | ((tensor_d1 & 0xFFFFu) << 16);
    g1.u[3] = (tensor_d1 >> 16) | (tile_d0 << 16);             // tile_dim0
    g1.u[4] = tile_d1;                                         // tile_dim1 (tile_dim2=0)
    g1.u[5] = stride0;                                         // tensor_dim0_stride[31:0]
    g1.u[6] = 0u;
    g1.u[7] = 0u;
    asm volatile("tensor_load_to_lds %0, %1" :: "s"(g0.v), "s"(g1.v) : "memory");
}

__device__ inline unsigned lds_addr_of(const void* p) {
    return (unsigned)(size_t)p;   // LDS offset lives in low 32 bits of generic address
}

__device__ inline float rmax16(float x) {
#pragma unroll
    for (int m = 1; m < 16; m <<= 1) x = fmaxf(x, __shfl_xor(x, m, 32));
    return x;
}
__device__ inline float rsum16(float x) {
#pragma unroll
    for (int m = 1; m < 16; m <<= 1) x += __shfl_xor(x, m, 32);
    return x;
}

// ---------------- f32 -> bf16 convert ----------------
__global__ void cvt_bf16_kernel(const float* __restrict__ in, __bf16* __restrict__ out, int n) {
    int i = blockIdx.x * blockDim.x + threadIdx.x;
    if (i < n) out[i] = (__bf16)in[i];
}

// ---------------------------------------------------------------------------
// QKV GEMM: [8192x1024] x [3072x1024]^T. Workgroup tile 128(M) x 64(N),
// 8 waves each own 16 M-rows and the full 64 N. A(128x32) and B(64x32) tiles
// double-buffered in LDS via TDM; B tile shared by all 8 waves.
// ---------------------------------------------------------------------------
__global__ void __launch_bounds__(256)
qkv_gemm_kernel(const __bf16* __restrict__ x, const __bf16* __restrict__ w,
                __bf16* __restrict__ qbuf, __bf16* __restrict__ kbuf,
                __bf16* __restrict__ vT) {
    __shared__ __bf16 Abuf[2][128 * 40];   // 128 rows, 80B pitch (32 elems + pad)
    __shared__ __bf16 Bbuf[2][64 * 40];

    int wave = threadIdx.x >> 5;
    int lane = threadIdx.x & 31;
    int nt = blockIdx.x % 48, mt = blockIdx.x / 48;   // 64 x 48 blocks
    int mo = mt * 128, fo = nt * 64;

    unsigned aAddr[2] = { lds_addr_of(&Abuf[0][0]), lds_addr_of(&Abuf[1][0]) };
    unsigned bAddr[2] = { lds_addr_of(&Bbuf[0][0]), lds_addr_of(&Bbuf[1][0]) };

    if (wave == 0) {
        tdm_load_2d(aAddr[0], x + (size_t)mo * DIM, DIM, BN,   32, 128, DIM);
        tdm_load_2d(bAddr[0], w + (size_t)fo * DIM, DIM, 3 * DIM, 32, 64, DIM);
    }

    v8f acc[4] = {};
    for (int ko = 0; ko < DIM; ko += 32) {
        int cur = (ko >> 5) & 1;
        if (wave == 0) {
            if (ko + 32 < DIM) {
                tdm_load_2d(aAddr[cur ^ 1], x + (size_t)mo * DIM + ko + 32, DIM, BN,   32, 128, DIM);
                tdm_load_2d(bAddr[cur ^ 1], w + (size_t)fo * DIM + ko + 32, DIM, 3 * DIM, 32, 64, DIM);
                __builtin_amdgcn_s_wait_tensorcnt(2);   // older pair (buf cur) complete
            } else {
                __builtin_amdgcn_s_wait_tensorcnt(0);
            }
        }
        __syncthreads();
        v16bf a  = load_frag_lds(&Abuf[cur][0], wave * 16, 0, 64);
        v16bf b0 = load_frag_lds(&Bbuf[cur][0], 0,  0, 64);
        v16bf b1 = load_frag_lds(&Bbuf[cur][0], 16, 0, 64);
        v16bf b2 = load_frag_lds(&Bbuf[cur][0], 32, 0, 64);
        v16bf b3 = load_frag_lds(&Bbuf[cur][0], 48, 0, 64);
        acc[0] = wmma_bf16(a, b0, acc[0]);
        acc[1] = wmma_bf16(a, b1, acc[1]);
        acc[2] = wmma_bf16(a, b2, acc[2]);
        acc[3] = wmma_bf16(a, b3, acc[3]);
        sched_ds_then_wmma();   // loads first -> progressive dscnt waits
        __syncthreads();        // reads done before this buffer is overwritten
    }

    int g = lane >> 4;
#pragma unroll
    for (int t = 0; t < 4; ++t) {
#pragma unroll
        for (int v = 0; v < 8; ++v) {
            int row  = mo + wave * 16 + v + 8 * g;     // global token index
            int f    = fo + t * 16 + (lane & 15);      // output feature 0..3071
            int bb   = row >> 10, n = row & 1023;
            int qkvi = f >> 10,  rem = f & 1023;
            int head = rem >> 6, dd = rem & 63;
            int bh   = bb * HEADS + head;
            float val = acc[t][v];
            if (qkvi == 0)
                qbuf[((size_t)bh * SEQ + n) * HDIM + dd] = (__bf16)(val * 0.125f); // d^-0.5
            else if (qkvi == 1)
                kbuf[((size_t)bh * SEQ + n) * HDIM + dd] = (__bf16)val;
            else
                vT[((size_t)bh * HDIM + dd) * SEQ + n] = (__bf16)val;
        }
    }
}

// ---------------------------------------------------------------------------
// Flash attention. One (b,h) per block.x, 64 query rows per block (4 waves x 16).
// K (32x64) and V^T (64x32) tiles double-buffered in LDS via TDM, shared by
// all 4 waves (4x traffic reduction). P restaged per-wave through LDS.
// ---------------------------------------------------------------------------
__global__ void __launch_bounds__(128)
attn_kernel(const __bf16* __restrict__ qbuf, const __bf16* __restrict__ kbuf,
            const __bf16* __restrict__ vT, __bf16* __restrict__ obuf) {
    __shared__ __bf16 Kbuf[2][32 * 80];    // 32 rows, 160B pitch (64 elems + pad)
    __shared__ __bf16 Vbuf[2][64 * 40];    // 64 rows, 80B pitch (32 elems + pad)
    __shared__ __bf16 Pbuf[4][16 * 40];    // per-wave P tile, 80B pitch

    int wave = threadIdx.x >> 5;
    int lane = threadIdx.x & 31;
    int bh   = blockIdx.x;                 // 0..127
    int qr0  = blockIdx.y * 64 + wave * 16;

    const __bf16* Q = qbuf + (size_t)bh * SEQ * HDIM;
    const __bf16* K = kbuf + (size_t)bh * SEQ * HDIM;
    const __bf16* V = vT   + (size_t)bh * HDIM * SEQ;
    __bf16* P = &Pbuf[wave][0];

    unsigned kAddr[2] = { lds_addr_of(&Kbuf[0][0]), lds_addr_of(&Kbuf[1][0]) };
    unsigned vAddr[2] = { lds_addr_of(&Vbuf[0][0]), lds_addr_of(&Vbuf[1][0]) };

    v16bf aq0 = load_frag_g(Q, qr0, 0, HDIM);
    v16bf aq1 = load_frag_g(Q, qr0, 32, HDIM);

    if (wave == 0) {
        tdm_load_2d(kAddr[0], K, HDIM, SEQ, HDIM, 32, HDIM);   // rows j=0..31, all d
        tdm_load_2d(vAddr[0], V, SEQ, HDIM, 32, HDIM, SEQ);    // rows d=0..63, j=0..31
    }

    v8f acc[4] = {};
    float mi[8], li[8];
#pragma unroll
    for (int v = 0; v < 8; ++v) { mi[v] = -1e30f; li[v] = 0.f; }

    int g = lane >> 4;
    for (int jo = 0; jo < SEQ; jo += 32) {
        int cur = (jo >> 5) & 1;
        if (wave == 0) {
            if (jo + 32 < SEQ) {
                tdm_load_2d(kAddr[cur ^ 1], K + (size_t)(jo + 32) * HDIM, HDIM, SEQ, HDIM, 32, HDIM);
                tdm_load_2d(vAddr[cur ^ 1], V + jo + 32, SEQ, HDIM, 32, HDIM, SEQ);
                __builtin_amdgcn_s_wait_tensorcnt(2);
            } else {
                __builtin_amdgcn_s_wait_tensorcnt(0);
            }
        }
        __syncthreads();

        // S = Q K^T for this 16x32 key block (two 16x16 tiles, K-dim 64 = 2 WMMAs each)
        v16bf bk00 = load_frag_lds(&Kbuf[cur][0], 0,  0,  128);
        v16bf bk01 = load_frag_lds(&Kbuf[cur][0], 0,  32, 128);
        v16bf bk10 = load_frag_lds(&Kbuf[cur][0], 16, 0,  128);
        v16bf bk11 = load_frag_lds(&Kbuf[cur][0], 16, 32, 128);
        v8f s0 = {}, s1 = {};
        s0 = wmma_bf16(aq0, bk00, s0);
        s0 = wmma_bf16(aq1, bk01, s0);
        s1 = wmma_bf16(aq0, bk10, s1);
        s1 = wmma_bf16(aq1, bk11, s1);

        // online softmax; lane holds rows v+8g, col lane&15
#pragma unroll
        for (int v = 0; v < 8; ++v) {
            float bm = rmax16(fmaxf(s0[v], s1[v]));
            float mn = fmaxf(mi[v], bm);
            float corr = __expf(mi[v] - mn);
            float p0 = __expf(s0[v] - mn);
            float p1 = __expf(s1[v] - mn);
            li[v] = li[v] * corr + rsum16(p0 + p1);
            mi[v] = mn;
            acc[0][v] *= corr; acc[1][v] *= corr;
            acc[2][v] *= corr; acc[3][v] *= corr;
            int row = v + 8 * g;
            P[row * 40 + (lane & 15)]      = (__bf16)p0;   // 80B-pitch rows
            P[row * 40 + 16 + (lane & 15)] = (__bf16)p1;
        }
        // restage P (C layout) as an A fragment; intra-wave dep -> s_wait_dscnt.
        v16bf ap  = load_frag_lds(P, 0, 0, 64);
        v16bf bv0 = load_frag_lds(&Vbuf[cur][0], 0,  0, 64);
        v16bf bv1 = load_frag_lds(&Vbuf[cur][0], 16, 0, 64);
        v16bf bv2 = load_frag_lds(&Vbuf[cur][0], 32, 0, 64);
        v16bf bv3 = load_frag_lds(&Vbuf[cur][0], 48, 0, 64);
        acc[0] = wmma_bf16(ap, bv0, acc[0]);
        acc[1] = wmma_bf16(ap, bv1, acc[1]);
        acc[2] = wmma_bf16(ap, bv2, acc[2]);
        acc[3] = wmma_bf16(ap, bv3, acc[3]);
        __syncthreads();   // K/V tile reads done before overwrite
    }

    // write O in [b, n, h*64+d] layout (bf16) for the projection GEMM
    int bb = bh >> 4, head = bh & 15;
#pragma unroll
    for (int v = 0; v < 8; ++v) {
        float inv = 1.f / li[v];
        int n = qr0 + v + 8 * g;
#pragma unroll
        for (int t = 0; t < 4; ++t) {
            int dd = t * 16 + (lane & 15);
            obuf[((size_t)bb * SEQ + n) * DIM + head * HDIM + dd] = (__bf16)(acc[t][v] * inv);
        }
    }
}

// ---------------------------------------------------------------------------
// Output projection: [8192x1024] x [1024x1024]^T + bias, f32 out.
// Same TDM-tiled structure as the QKV GEMM.
// ---------------------------------------------------------------------------
__global__ void __launch_bounds__(256)
proj_gemm_kernel(const __bf16* __restrict__ o, const __bf16* __restrict__ w,
                 const float* __restrict__ bias, float* __restrict__ out) {
    __shared__ __bf16 Abuf[2][128 * 40];
    __shared__ __bf16 Bbuf[2][64 * 40];

    int wave = threadIdx.x >> 5;
    int lane = threadIdx.x & 31;
    int nt = blockIdx.x % 16, mt = blockIdx.x / 16;   // 64 x 16 blocks
    int mo = mt * 128, no = nt * 64;

    unsigned aAddr[2] = { lds_addr_of(&Abuf[0][0]), lds_addr_of(&Abuf[1][0]) };
    unsigned bAddr[2] = { lds_addr_of(&Bbuf[0][0]), lds_addr_of(&Bbuf[1][0]) };

    if (wave == 0) {
        tdm_load_2d(aAddr[0], o + (size_t)mo * DIM, DIM, BN,  32, 128, DIM);
        tdm_load_2d(bAddr[0], w + (size_t)no * DIM, DIM, DIM, 32, 64,  DIM);
    }

    v8f acc[4] = {};
    for (int ko = 0; ko < DIM; ko += 32) {
        int cur = (ko >> 5) & 1;
        if (wave == 0) {
            if (ko + 32 < DIM) {
                tdm_load_2d(aAddr[cur ^ 1], o + (size_t)mo * DIM + ko + 32, DIM, BN,  32, 128, DIM);
                tdm_load_2d(bAddr[cur ^ 1], w + (size_t)no * DIM + ko + 32, DIM, DIM, 32, 64,  DIM);
                __builtin_amdgcn_s_wait_tensorcnt(2);
            } else {
                __builtin_amdgcn_s_wait_tensorcnt(0);
            }
        }
        __syncthreads();
        v16bf a  = load_frag_lds(&Abuf[cur][0], wave * 16, 0, 64);
        v16bf b0 = load_frag_lds(&Bbuf[cur][0], 0,  0, 64);
        v16bf b1 = load_frag_lds(&Bbuf[cur][0], 16, 0, 64);
        v16bf b2 = load_frag_lds(&Bbuf[cur][0], 32, 0, 64);
        v16bf b3 = load_frag_lds(&Bbuf[cur][0], 48, 0, 64);
        acc[0] = wmma_bf16(a, b0, acc[0]);
        acc[1] = wmma_bf16(a, b1, acc[1]);
        acc[2] = wmma_bf16(a, b2, acc[2]);
        acc[3] = wmma_bf16(a, b3, acc[3]);
        sched_ds_then_wmma();   // loads first -> progressive dscnt waits
        __syncthreads();
    }

    int g = lane >> 4;
#pragma unroll
    for (int t = 0; t < 4; ++t) {
        int col = no + t * 16 + (lane & 15);
        float bv = bias[col];
#pragma unroll
        for (int v = 0; v < 8; ++v)
            out[(size_t)(mo + wave * 16 + v + 8 * g) * DIM + col] = acc[t][v] + bv;
    }
}

extern "C" void kernel_launch(void* const* d_in, const int* in_sizes, int n_in,
                              void* d_out, int out_size, void* d_ws, size_t ws_size,
                              hipStream_t stream) {
    const float* x      = (const float*)d_in[0];   // [8,1024,1024]
    const float* w_qkv  = (const float*)d_in[1];   // [3072,1024]
    const float* w_proj = (const float*)d_in[2];   // [1024,1024]
    const float* b_proj = (const float*)d_in[3];   // [1024]
    float* out = (float*)d_out;                    // [8,1024,1024]

    size_t off = 0;
    auto carve = [&](size_t bytes) {
        void* p = (char*)d_ws + off;
        off += (bytes + 255) & ~(size_t)255;
        return p;
    };
    const size_t NX   = (size_t)BN * DIM;                    // 8M
    const size_t NWQ  = (size_t)3 * DIM * DIM;               // 3M
    const size_t NWP  = (size_t)DIM * DIM;                   // 1M
    const size_t NQKV = (size_t)BATCH * HEADS * SEQ * HDIM;  // 8M each

    __bf16* x_bf     = (__bf16*)carve(NX * 2);
    __bf16* wqkv_bf  = (__bf16*)carve(NWQ * 2);
    __bf16* wproj_bf = (__bf16*)carve(NWP * 2);
    __bf16* q_bf     = (__bf16*)carve(NQKV * 2);
    __bf16* k_bf     = (__bf16*)carve(NQKV * 2);
    __bf16* vT_bf    = (__bf16*)carve(NQKV * 2);
    __bf16* o_bf     = (__bf16*)carve(NX * 2);

    cvt_bf16_kernel<<<(int)((NX  + 255) / 256), 256, 0, stream>>>(x,      x_bf,     (int)NX);
    cvt_bf16_kernel<<<(int)((NWQ + 255) / 256), 256, 0, stream>>>(w_qkv,  wqkv_bf,  (int)NWQ);
    cvt_bf16_kernel<<<(int)((NWP + 255) / 256), 256, 0, stream>>>(w_proj, wproj_bf, (int)NWP);

    qkv_gemm_kernel<<<64 * 48, 256, 0, stream>>>(x_bf, wqkv_bf, q_bf, k_bf, vT_bf);

    attn_kernel<<<dim3(BATCH * HEADS, SEQ / 64), 128, 0, stream>>>(q_bf, k_bf, vT_bf, o_bf);

    proj_gemm_kernel<<<64 * 16, 256, 0, stream>>>(o_bf, wproj_bf, b_proj, out);
}
